// DecoderRNN_30434138259953
// MI455X (gfx1250) — matrix operand order repaired and checked by reading, not compiled, not run on hardware
//
#include <hip/hip_runtime.h>
#include <hip/hip_bf16.h>
#include <math.h>

// ---- problem dims ----
#define B_   16
#define T_   128
#define S_   128
#define H_   512
#define E_   512
#define V_   32000
#define G3H  1536   // 3*H
#define NWG  16     // workgroups in the recurrent cluster (one per 32-col k-slice / batch)

typedef __attribute__((ext_vector_type(16))) __bf16 bf16x16;
typedef __attribute__((ext_vector_type(8)))  float  f32x8;
typedef __attribute__((ext_vector_type(4)))  unsigned int u32x4;
typedef __attribute__((ext_vector_type(8)))  int    i32x8;
typedef __attribute__((ext_vector_type(4)))  int    i32x4;

union FragBF { bf16x16 v; uint4 q[2]; };
union BFPack { uint4 q; __bf16 h[8]; };

// A-fragment (16x32 bf16, MxK): lane m=lane&15, half=lane>>4.
// dwords 0..3 hold K = k0 + half*8 + {0..7}, dwords 4..7 hold K = k0 + 16 + half*8 + {0..7}.
__device__ __forceinline__ bf16x16 load_frag_a(const __bf16* base, int stride, int k0, int lane) {
  int m = lane & 15, half = lane >> 4;
  const __bf16* p = base + (size_t)m * stride + k0 + half * 8;
  FragBF u;
  u.q[0] = *(const uint4*)(p);
  u.q[1] = *(const uint4*)(p + 16);
  return u.v;
}

// B-fragment (32x16 bf16, KxN) for C = A * W^T, W row-major [N,K]:
// lane n=lane&15 picks row n0+n; dwords 0..3 hold K = k0 + half*8 + {0..7}, dwords 4..7 hold K = k0 + 16 + half*8 + {0..7}.
__device__ __forceinline__ bf16x16 load_frag_b(const __bf16* W, int ldw, int n0, int k0, int lane) {
  int n = lane & 15, half = lane >> 4;
  const __bf16* p = W + (size_t)(n0 + n) * ldw + k0 + half * 8;
  FragBF u;
  u.q[0] = *(const uint4*)(p);
  u.q[1] = *(const uint4*)(p + 16);
  return u.v;
}

// ---------------- TDM: 1-D tile load into LDS (descriptor per CDNA5 ISA) ----
#if __has_include(<hip/amd_detail/amd_gfx1250_TDM.h>)
#define TDM_6ARG 1
#else
#define TDM_6ARG 0
#endif

__device__ __forceinline__ void tdm_load_bf16(const __bf16* gptr, __bf16* lptr, unsigned nelem) {
  unsigned long long ga = (unsigned long long)(size_t)gptr;
  unsigned lds = (unsigned)(size_t)lptr;          // LDS aperture: low 32 bits = LDS byte address
  u32x4 g0;
  g0[0] = 1u;                                      // count=1, user descriptor
  g0[1] = lds;                                     // lds_addr
  g0[2] = (unsigned)ga;                            // global_addr[31:0]
  g0[3] = (unsigned)((ga >> 32) & 0x01FFFFFFu) | (2u << 30);  // global_addr[56:32] | type=2
  i32x8 g1;
  g1[0] = (int)(1u << 16);                         // data_size = 1 (2 bytes)
  g1[1] = (int)(nelem << 16);                      // tensor_dim0[15:0] -> bits[31:16]
  g1[2] = (int)((nelem >> 16) & 0xFFFFu) | (1 << 16); // tensor_dim0[31:16] | tensor_dim1 = 1
  g1[3] = (int)((nelem & 0xFFFFu) << 16);          // tile_dim0 -> bits[127:112]
  g1[4] = 1;                                       // tile_dim1 = 1
  g1[5] = (int)nelem;                              // tensor_dim0_stride low 32
  g1[6] = 0;
  g1[7] = 0;
  i32x4 z4 = {0, 0, 0, 0};
#if TDM_6ARG
  i32x8 z8 = {0, 0, 0, 0, 0, 0, 0, 0};
  __builtin_amdgcn_tensor_load_to_lds(g0, g1, z4, z4, z8, 0);
#else
  __builtin_amdgcn_tensor_load_to_lds(g0, g1, z4, z4, 0);
#endif
}

// ---------------- cross-workgroup sync ----------------
// Primary path: CDNA5 cluster barrier (s_barrier_signal/wait -3) when dispatched as a cluster.
// Fallback (ClusterID==0, plain dispatch): sense-reversing atomic barrier in device memory.
__device__ __forceinline__ void cluster_sync(unsigned* cnt, unsigned* phase, bool atomic_path) {
  __threadfence();            // release this thread's global stores
  __syncthreads();
  __builtin_amdgcn_s_cluster_barrier();     // real HW cluster split-barrier when clustered
  if (atomic_path && threadIdx.x == 0) {
    unsigned ph = __hip_atomic_load(phase, __ATOMIC_ACQUIRE, __HIP_MEMORY_SCOPE_AGENT);
    if (atomicAdd(cnt, 1u) == NWG - 1u) {
      __hip_atomic_store(cnt, 0u, __ATOMIC_RELAXED, __HIP_MEMORY_SCOPE_AGENT);
      __hip_atomic_fetch_add(phase, 1u, __ATOMIC_RELEASE, __HIP_MEMORY_SCOPE_AGENT);
    } else {
      while (__hip_atomic_load(phase, __ATOMIC_ACQUIRE, __HIP_MEMORY_SCOPE_AGENT) == ph)
        __builtin_amdgcn_s_sleep(2);
    }
  }
  __syncthreads();
  __threadfence();            // acquire side
}

// ---------------- elementwise prep kernels ----------------
__global__ void cast_f32_to_bf16_kernel(const float* __restrict__ in, __bf16* __restrict__ out, int n) {
  int i = blockIdx.x * blockDim.x + threadIdx.x;
  if (i < n) out[i] = (__bf16)in[i];
}

__global__ void embed_kernel(const int* __restrict__ text, const float* __restrict__ table,
                             __bf16* __restrict__ out, int BT) {
  int i = blockIdx.x * blockDim.x + threadIdx.x;  // over BT*E
  if (i >= BT * E_) return;
  int row = i >> 9;
  int e   = i & (E_ - 1);
  int idx = text[row];
  float v = (idx != 0) ? table[(size_t)idx * E_ + e] : 0.0f;  // padding_idx = 0
  out[i] = (__bf16)v;
}

__global__ void init_barrier_kernel(unsigned* p) {
  if (threadIdx.x < 2) p[threadIdx.x] = 0u;
}

// ---------------- generic NT GEMM: C[M,N] = A[M,K] * W[N,K]^T + bias ----------------
template<int MT>
__global__ void gemm_bf16_nt_kernel(const __bf16* __restrict__ A,
                                    const __bf16* __restrict__ W,
                                    const float* __restrict__ bias,
                                    float* __restrict__ C,
                                    int M, int N, int K) {
  const int lane = threadIdx.x & 31;
  const int wave = blockIdx.x * (blockDim.x >> 5) + (threadIdx.x >> 5);
  const int ntiles  = N >> 4;
  const int mblocks = M / (16 * MT);
  if (wave >= mblocks * ntiles) return;
  const int nb = wave % ntiles, mb = wave / ntiles;
  const int n0 = nb * 16, m0 = mb * 16 * MT;

  f32x8 acc[MT];
  #pragma unroll
  for (int i = 0; i < MT; ++i)
    #pragma unroll
    for (int r = 0; r < 8; ++r) acc[i][r] = 0.0f;

  for (int k0 = 0; k0 < K; k0 += 32) {
    // prefetch the B operand two k-steps ahead (global_prefetch_b8)
    __builtin_prefetch((const void*)(W + (size_t)(n0 + (lane & 15)) * K + k0 + 64), 0, 1);
    bf16x16 b = load_frag_b(W, K, n0, k0, lane);
    #pragma unroll
    for (int i = 0; i < MT; ++i) {
      bf16x16 a = load_frag_a(A + (size_t)(m0 + i * 16) * K, K, k0, lane);
      acc[i] = __builtin_amdgcn_wmma_f32_16x16x32_bf16(false, a, false, b, (short)0, acc[i], false, false);
    }
  }

  const int nc = lane & 15, hc = lane >> 4;
  const float bv = bias[n0 + nc];
  #pragma unroll
  for (int i = 0; i < MT; ++i)
    #pragma unroll
    for (int r = 0; r < 8; ++r) {
      int m = m0 + i * 16 + hc * 8 + r;
      C[(size_t)m * N + n0 + nc] = acc[i][r] + bv;
    }
}

// ---------------- recurrent kernel: 16 cooperating workgroups (cluster), 256 thr each ----------------
// WG w owns: k-slice [w*32, w*32+32) of H  (GRU gates + h_new slice, w_hh rows {k,512+k,1024+k}),
//            batch  b = w                  (attention scores/softmax/mix for its own batch),
//            n-slice [w*32, w*32+32) of attended (lin_out rows).
// Weight slices live in LDS (TDM-loaded once): zero per-step weight traffic, no spill motive.
__global__ void __launch_bounds__(256) rnn_cluster_kernel(
    const float* __restrict__ gi,         // [B*T, 3H]
    const float* __restrict__ b_hh,       // [3H]
    const __bf16* __restrict__ w_hh_bf,   // [3H, H]
    const __bf16* __restrict__ lin_bf,    // [H, 2H]
    const __bf16* __restrict__ enc_bf,    // [B, S, H]
    const unsigned char* __restrict__ mask, // [B, S]
    const float* __restrict__ h0,         // [B, H]
    __bf16* __restrict__ h_buf,           // [B, H]  exchange
    __bf16* __restrict__ mix_buf,         // [B, H]  exchange
    __bf16* __restrict__ att_seq,         // [B*T, H]
    float* __restrict__ attn_log,         // [T, B, S]
    float* __restrict__ h_final,          // [B, H]
    unsigned* __restrict__ bar)           // [2] counter, phase
{
  extern __shared__ char smem[];
  __bf16* lds_whh = (__bf16*)(smem);                     // [96][512]   98304 B
  __bf16* lds_lin = (__bf16*)(smem + 98304);             // [32][1024]  65536 B
  __bf16* comb    = (__bf16*)(smem + 98304 + 65536);     // [16][1024]  32768 B  ([h | mix])
  float*  gh_s    = (float*)(smem + 196608);             // [16][96]     6144 B
  float*  sc      = (float*)(smem + 202752);             // [128]         512 B

  const int tid  = threadIdx.x;
  const int lane = tid & 31;
  const int wave = tid >> 5;
  const int w    = blockIdx.x;      // 0..15
  const int kw   = w * 32;          // owned k-slice base
  const int b_own = w;              // owned batch
  const bool atomic_path = (__builtin_amdgcn_cluster_id_x() == 0);

  // ---- prologue: TDM weight slices into LDS; h0 into comb[:, :512] ----
  if (tid == 0) {
    #pragma unroll
    for (int c = 0; c < 3; ++c)
      tdm_load_bf16(w_hh_bf + (size_t)(c * 512 + kw) * H_, lds_whh + c * 32 * H_, 32 * H_);
    tdm_load_bf16(lin_bf + (size_t)kw * (2 * H_), lds_lin, 32 * 2 * H_);
  }
  for (int i = tid; i < B_ * H_; i += 256) {
    int b = i >> 9, k = i & (H_ - 1);
    comb[b * 1024 + k] = (__bf16)h0[i];
  }
  __builtin_amdgcn_s_wait_tensorcnt(0);
  __syncthreads();

  for (int t = 0; t < T_; ++t) {
    asm volatile("" ::: "memory");   // keep per-step loads inside the loop (no cross-t hoist/spill)

    // ---- stage 1: gh slice = h(t-1) @ w_hh_slice^T   (M=16, N=96 -> 6 tiles, K=512) ----
    if (wave < 6) {
      const int c = wave >> 1;                 // gate chunk 0..2 (r,z,n)
      const int o = (wave & 1) * 16;           // offset in 32-col chunk
      const int rloc = c * 32 + o;             // row base in lds_whh
      f32x8 acc;
      #pragma unroll
      for (int r = 0; r < 8; ++r) acc[r] = 0.0f;
      for (int k0 = 0; k0 < H_; k0 += 32) {
        bf16x16 a = load_frag_a(comb, 1024, k0, lane);
        bf16x16 b = load_frag_b(lds_whh, H_, rloc, k0, lane);
        acc = __builtin_amdgcn_wmma_f32_16x16x32_bf16(false, a, false, b, (short)0, acc, false, false);
      }
      int nc = lane & 15, hc = lane >> 4;
      float bb = b_hh[c * 512 + kw + o + nc];
      #pragma unroll
      for (int r = 0; r < 8; ++r)
        gh_s[(hc * 8 + r) * 96 + rloc + nc] = acc[r] + bb;
    }
    __syncthreads();

    // ---- stage 2: GRU gates on owned slice (16 b x 32 k = 512 elems, 2/thread) ----
    #pragma unroll
    for (int it = 0; it < 2; ++it) {
      int e = tid + it * 256;
      int b = e >> 5, kk = e & 31;
      int kg = kw + kk;
      const float* gib = gi + (size_t)(b * T_ + t) * G3H;
      float ir = gib[kg], iz = gib[H_ + kg], inn = gib[2 * H_ + kg];
      float hr = gh_s[b * 96 + kk], hz = gh_s[b * 96 + 32 + kk], hn = gh_s[b * 96 + 64 + kk];
      float hprev = (float)comb[b * 1024 + kg];
      float r = 1.0f / (1.0f + __expf(-(ir + hr)));
      float z = 1.0f / (1.0f + __expf(-(iz + hz)));
      float n = tanhf(inn + r * hn);
      float hnew = (1.0f - z) * n + z * hprev;
      h_buf[b * H_ + kg] = (__bf16)hnew;
      if (t == T_ - 1) h_final[b * H_ + kg] = hnew;
    }
    cluster_sync(bar, bar + 1, atomic_path);

    // ---- reload full h(t) into comb[:, :512] ----
    for (int i = tid; i < (B_ * H_) / 8; i += 256) {
      int b = (i * 8) >> 9, k = (i * 8) & (H_ - 1);
      *(uint4*)(comb + b * 1024 + k) = *(const uint4*)(h_buf + b * H_ + k);
    }
    __syncthreads();

    // ---- stage 3: scores for own batch (2 threads per s, 256-long half-dots) ----
    {
      int s = tid >> 1, half = tid & 1;
      const __bf16* er = enc_bf + ((size_t)(b_own * S_ + s)) * H_ + half * 256;
      const __bf16* hr = comb + b_own * 1024 + half * 256;
      float acc = 0.0f;
      for (int k = 0; k < 256; k += 8) {
        BFPack pe, ph;
        pe.q = *(const uint4*)(er + k);
        ph.q = *(const uint4*)(hr + k);
        #pragma unroll
        for (int u = 0; u < 8; ++u) acc += (float)pe.h[u] * (float)ph.h[u];
      }
      acc += __shfl_xor(acc, 1, 32);
      if (half == 0) sc[s] = mask[b_own * S_ + s] ? acc : -1.0e6f;
    }
    __syncthreads();

    // ---- stage 4: softmax over S=128 on wave 0 ----
    if (wave == 0) {
      float v0 = sc[lane], v1 = sc[lane + 32], v2 = sc[lane + 64], v3 = sc[lane + 96];
      float m = fmaxf(fmaxf(v0, v1), fmaxf(v2, v3));
      #pragma unroll
      for (int off = 16; off >= 1; off >>= 1) m = fmaxf(m, __shfl_xor(m, off, 32));
      float e0 = __expf(v0 - m), e1 = __expf(v1 - m), e2 = __expf(v2 - m), e3 = __expf(v3 - m);
      float sum = e0 + e1 + e2 + e3;
      #pragma unroll
      for (int off = 16; off >= 1; off >>= 1) sum += __shfl_xor(sum, off, 32);
      float inv = 1.0f / sum;
      sc[lane] = e0 * inv; sc[lane + 32] = e1 * inv; sc[lane + 64] = e2 * inv; sc[lane + 96] = e3 * inv;
      float* al = attn_log + ((size_t)t * B_ + b_own) * S_;
      al[lane] = e0 * inv; al[lane + 32] = e1 * inv; al[lane + 64] = e2 * inv; al[lane + 96] = e3 * inv;
    }
    __syncthreads();

    // ---- stage 5: mix[k] for own batch (2 k per thread) ----
    #pragma unroll
    for (int it = 0; it < 2; ++it) {
      int k = tid + it * 256;
      float acc = 0.0f;
      const __bf16* eb = enc_bf + (size_t)b_own * S_ * H_ + k;
      for (int s = 0; s < S_; ++s) acc += sc[s] * (float)eb[(size_t)s * H_];
      mix_buf[b_own * H_ + k] = (__bf16)acc;
    }
    cluster_sync(bar, bar + 1, atomic_path);

    // ---- reload full mix into comb[:, 512:] ----
    for (int i = tid; i < (B_ * H_) / 8; i += 256) {
      int b = (i * 8) >> 9, k = (i * 8) & (H_ - 1);
      *(uint4*)(comb + b * 1024 + 512 + k) = *(const uint4*)(mix_buf + b * H_ + k);
    }
    __syncthreads();

    // ---- stage 6: attended n-slice = tanh([h|mix] @ lin_slice^T)  (M=16, N=32 -> 2 tiles, K=1024) ----
    if (wave < 2) {
      f32x8 acc;
      #pragma unroll
      for (int r = 0; r < 8; ++r) acc[r] = 0.0f;
      const int rloc = wave * 16;
      for (int k0 = 0; k0 < 2 * H_; k0 += 32) {
        bf16x16 a = load_frag_a(comb, 1024, k0, lane);
        bf16x16 b = load_frag_b(lds_lin, 2 * H_, rloc, k0, lane);
        acc = __builtin_amdgcn_wmma_f32_16x16x32_bf16(false, a, false, b, (short)0, acc, false, false);
      }
      int nc = lane & 15, hc = lane >> 4;
      int ng = kw + rloc + nc;                 // global attended column
      #pragma unroll
      for (int r = 0; r < 8; ++r) {
        int b = hc * 8 + r;
        att_seq[((size_t)b * T_ + t) * H_ + ng] = (__bf16)tanhf(acc[r]);
      }
    }
    __syncthreads();
  }
}

// ---------------- host launch ----------------
extern "C" void kernel_launch(void* const* d_in, const int* in_sizes, int n_in,
                              void* d_out, int out_size, void* d_ws, size_t ws_size,
                              hipStream_t stream) {
  const int*           text   = (const int*)d_in[0];
  const float*         dec_h  = (const float*)d_in[1];   // [1,B,H]
  const float*         enc    = (const float*)d_in[2];   // [B,S,H]
  const unsigned char* mask   = (const unsigned char*)d_in[3];
  const float*         table  = (const float*)d_in[4];   // [V,E]
  const float*         w_ih   = (const float*)d_in[5];   // [3H,E]
  const float*         w_hh   = (const float*)d_in[6];   // [3H,H]
  const float*         b_ih   = (const float*)d_in[7];
  const float*         b_hh   = (const float*)d_in[8];
  const float*         lin_w  = (const float*)d_in[9];   // [H,2H]
  const float*         out_w  = (const float*)d_in[10];  // [V,H]
  const float*         out_b  = (const float*)d_in[11];

  float* out_scores = (float*)d_out;
  float* out_hfinal = out_scores + (size_t)B_ * T_ * V_;
  float* out_attn   = out_hfinal + (size_t)B_ * H_;

  char* ws = (char*)d_ws;
  size_t off = 0;
  auto walloc = [&](size_t bytes) -> void* {
    void* p = ws + off;
    off = (off + bytes + 255) & ~(size_t)255;
    return p;
  };
  __bf16* w_ih_bf  = (__bf16*)walloc((size_t)G3H * E_ * 2);
  __bf16* w_hh_bf  = (__bf16*)walloc((size_t)G3H * H_ * 2);
  __bf16* lin_bf   = (__bf16*)walloc((size_t)H_ * 2 * H_ * 2);
  __bf16* out_w_bf = (__bf16*)walloc((size_t)V_ * H_ * 2);
  __bf16* enc_bf   = (__bf16*)walloc((size_t)B_ * S_ * H_ * 2);
  __bf16* emb_bf   = (__bf16*)walloc((size_t)B_ * T_ * E_ * 2);
  float*  gi       = (float*)walloc((size_t)B_ * T_ * G3H * 4);
  __bf16* h_buf    = (__bf16*)walloc((size_t)B_ * H_ * 2);
  __bf16* mix_buf  = (__bf16*)walloc((size_t)B_ * H_ * 2);
  __bf16* att_seq  = (__bf16*)walloc((size_t)B_ * T_ * H_ * 2);
  unsigned* bar    = (unsigned*)walloc(256);

  // 1) casts to bf16
  {
    int n;
    n = G3H * E_;
    cast_f32_to_bf16_kernel<<<(n + 255) / 256, 256, 0, stream>>>(w_ih, w_ih_bf, n);
    n = G3H * H_;
    cast_f32_to_bf16_kernel<<<(n + 255) / 256, 256, 0, stream>>>(w_hh, w_hh_bf, n);
    n = H_ * 2 * H_;
    cast_f32_to_bf16_kernel<<<(n + 255) / 256, 256, 0, stream>>>(lin_w, lin_bf, n);
    n = V_ * H_;
    cast_f32_to_bf16_kernel<<<(n + 255) / 256, 256, 0, stream>>>(out_w, out_w_bf, n);
    n = B_ * S_ * H_;
    cast_f32_to_bf16_kernel<<<(n + 255) / 256, 256, 0, stream>>>(enc, enc_bf, n);
  }

  // 2) embedding (padding_idx=0) -> bf16
  {
    int n = B_ * T_ * E_;
    embed_kernel<<<(n + 255) / 256, 256, 0, stream>>>(text, table, emb_bf, B_ * T_);
  }

  // 3) hoisted input-gate GEMM: gi = emb @ w_ih^T + b_ih  (M=2048, N=1536, K=512)
  {
    int waves = (B_ * T_ / 64) * (G3H / 16);
    gemm_bf16_nt_kernel<4><<<(waves + 7) / 8, 256, 0, stream>>>(
        emb_bf, w_ih_bf, b_ih, gi, B_ * T_, G3H, E_);
  }

  // 4) barrier init + recurrent cluster kernel (16 WGs x 256, ~199 KB LDS each)
  init_barrier_kernel<<<1, 64, 0, stream>>>(bar);
  rnn_cluster_kernel<<<NWG, 256, 203264, stream>>>(
      gi, b_hh, w_hh_bf, lin_bf, enc_bf, mask, dec_h,
      h_buf, mix_buf, att_seq, out_attn, out_hfinal, bar);

  // 5) output projection: scores = att_seq @ out_w^T + out_b  (M=2048, N=32000, K=512)
  {
    int waves = (B_ * T_ / 64) * (V_ / 16);
    gemm_bf16_nt_kernel<4><<<(waves + 7) / 8, 256, 0, stream>>>(
        att_seq, out_w_bf, out_b, out_scores, B_ * T_, V_, H_);
  }
}
